// BotDetectionTransformer_83863531421899
// MI455X (gfx1250) — matrix operand order repaired and checked by reading, not compiled, not run on hardware
//
#include <hip/hip_runtime.h>

// ---------------------------------------------------------------------------
// BotDetectionTransformer for MI455X (gfx1250, wave32, WMMA + async-to-LDS).
// All GEMMs + both attention einsums run on v_wmma_f32_16x16x32_bf16
// (f32 accumulate).  Tile staging uses GLOBAL_LOAD_ASYNC_TO_LDS_B128
// (ASYNCcnt) with double-buffered LDS.  Softmax / layernorm in fp32 VALU.
// ---------------------------------------------------------------------------

typedef unsigned short u16;
typedef __attribute__((ext_vector_type(16))) __bf16 bf16x16;
typedef __attribute__((ext_vector_type(8)))  float  f32x8;

#define Bsz  8
#define Ssz  1024
#define Dm   768
#define Hn   12
#define HDm  64
#define DFm  3072
#define Ln   6
#define EPSC 1e-5f

union FragBF {            // 32 bytes = one 16-element bf16 WMMA operand
  bf16x16 v;
  uint4   u[2];
};

__device__ __forceinline__ u16 f2bf(float f) {           // RNE f32 -> bf16
  union { float f; unsigned u; } a; a.f = f;
  unsigned u = a.u + 0x7fffu + ((a.u >> 16) & 1u);
  return (u16)(u >> 16);
}
__device__ __forceinline__ float bf2f(u16 h) {
  union { unsigned u; float f; } a; a.u = ((unsigned)h) << 16;
  return a.f;
}

__device__ __forceinline__ f32x8 wmma_bf16(const bf16x16 a, const bf16x16 b,
                                           const f32x8 c) {
  return __builtin_amdgcn_wmma_f32_16x16x32_bf16(
      false, a, false, b, (short)0, c, false, false);
}

// Generic pointers to __shared__ are {SHARED_BASE_hi32, lds_offset_lo32}
// (ISA 10.2 aperture rules), so the low 32 bits are the per-lane LDS address
// the async-load VDST operand wants (HW adds LDS_BASE).
__device__ __forceinline__ unsigned lds_addr(const void* p) {
  return (unsigned)(unsigned long long)p;
}
__device__ __forceinline__ void async_copy_b128(unsigned lds, const void* g) {
  asm volatile("global_load_async_to_lds_b128 %0, %1, off"
               :: "v"(lds), "v"((unsigned long long)g) : "memory");
}
__device__ __forceinline__ void wait_async() {
  asm volatile("s_wait_asynccnt 0x0" ::: "memory");
}

// ---------------------------------------------------------------------------
// Embedding: x = (token_emb[id] + pos_emb[s]) * sqrt(D)
// ---------------------------------------------------------------------------
__global__ __launch_bounds__(256)
void embed_kernel(const int* __restrict__ ids, const float* __restrict__ tok,
                  const float* __restrict__ pos, float* __restrict__ x) {
  const int bs = blockIdx.x;            // 0 .. B*S-1
  const int s  = bs & (Ssz - 1);
  const int id = ids[bs];
  const float sc = 27.712812921102035f; // sqrt(768)
  const float* tr = tok + (size_t)id * Dm;
  const float* pr = pos + (size_t)s  * Dm;
  float* xr = x + (size_t)bs * Dm;
#pragma unroll
  for (int j = 0; j < 3; ++j) {
    int d = threadIdx.x + j * 256;
    xr[d] = (tr[d] + pr[d]) * sc;
  }
}

// ---------------------------------------------------------------------------
// f32 -> bf16 weight conversion
// ---------------------------------------------------------------------------
__global__ __launch_bounds__(256)
void convert_kernel(const float* __restrict__ in, u16* __restrict__ out, int n) {
  for (int i = blockIdx.x * 256 + threadIdx.x; i < n; i += gridDim.x * 256)
    out[i] = f2bf(in[i]);
}

// ---------------------------------------------------------------------------
// Block reduce (256 threads = 8 waves of 32)
// ---------------------------------------------------------------------------
__device__ __forceinline__ float block_sum(float v, float* red) {
#pragma unroll
  for (int off = 16; off > 0; off >>= 1) v += __shfl_xor(v, off, 32);
  const int wid = threadIdx.x >> 5, lane = threadIdx.x & 31;
  if (lane == 0) red[wid] = v;
  __syncthreads();
  if (wid == 0) {
    float t = (lane < 8) ? red[lane] : 0.f;
#pragma unroll
    for (int off = 4; off > 0; off >>= 1) t += __shfl_xor(t, off, 32);
    if (lane == 0) red[0] = t;
  }
  __syncthreads();
  float r = red[0];
  __syncthreads();
  return r;
}

// ---------------------------------------------------------------------------
// LayerNorm over D=768, writes bf16 activations for the next GEMM
// ---------------------------------------------------------------------------
__global__ __launch_bounds__(256)
void layernorm_kernel(const float* __restrict__ x, const float* __restrict__ sc,
                      const float* __restrict__ bi, u16* __restrict__ out) {
  __shared__ float red[8];
  const int row = blockIdx.x;
  const float* xr = x + (size_t)row * Dm;
  const int t = threadIdx.x;
  float v0 = xr[t], v1 = xr[t + 256], v2 = xr[t + 512];
  float mean = block_sum(v0 + v1 + v2, red) * (1.f / Dm);
  float d0 = v0 - mean, d1 = v1 - mean, d2 = v2 - mean;
  float var = block_sum(d0 * d0 + d1 * d1 + d2 * d2, red) * (1.f / Dm);
  float inv = rsqrtf(var + EPSC);
  u16* orow = out + (size_t)row * Dm;
  orow[t]       = f2bf(d0 * inv * sc[t]       + bi[t]);
  orow[t + 256] = f2bf(d1 * inv * sc[t + 256] + bi[t + 256]);
  orow[t + 512] = f2bf(d2 * inv * sc[t + 512] + bi[t + 512]);
}

// ---------------------------------------------------------------------------
// bf16 WMMA GEMM:  D[M,N] = A[M,K] @ B[K,N] + bias (+resid) (ReLU opt)
// Block tile 128x128, 8 waves as 4x2, wave tile 32x64 (2x4 WMMA), K step 32.
// Double-buffered LDS; A tile staged with async-to-LDS (ASYNCcnt), B tile
// transposed through VGPRs (needed for the B-fragment layout K = lh*8+e / 16+lh*8+(e-8)).
// ---------------------------------------------------------------------------
#define BM  128
#define BN  128
#define LDA 40   // padded LDS strides (elements); byte stride mult of 16
#define LDB 40

__global__ __launch_bounds__(256)
void gemm_bf16_kernel(const u16* __restrict__ A, const u16* __restrict__ Bw,
                      const float* __restrict__ bias,
                      const float* __restrict__ resid,
                      float* __restrict__ outF, u16* __restrict__ outB,
                      int M, int N, int K, int relu) {
  __shared__ u16 lA[2][BM * LDA];
  __shared__ u16 lB[2][BN * LDB];
  const int tid  = threadIdx.x;
  const int wid  = tid >> 5, lane = tid & 31;
  const int lh   = lane >> 4, l16 = lane & 15;
  const int wm   = wid >> 1,  wn  = wid & 1;
  const int bm0  = blockIdx.y * BM, bn0 = blockIdx.x * BN;

  // staging helpers -------------------------------------------------------
  auto stageA = [&](int buf, int k0) {       // 128 rows, 16B/thread async (k0+0..7 / k0+16..23)
    const int row = tid >> 1, half = tid & 1;
    async_copy_b128(lds_addr(&lA[buf][row * LDA + half * 16]),
                    A + (size_t)(bm0 + row) * K + k0 + half * 16);
  };
  auto stageB = [&](int buf, int k0) {       // 32x128 -> lB[n][k] transpose
    const int kr = tid >> 3, n0 = (tid & 7) * 16;
    const uint4* g = (const uint4*)(Bw + (size_t)(k0 + kr) * N + bn0 + n0);
    union { uint4 u; u16 s[8]; } d0, d1;
    d0.u = g[0]; d1.u = g[1];
#pragma unroll
    for (int j = 0; j < 8; ++j) {
      lB[buf][(n0 + j) * LDB + kr]     = d0.s[j];
      lB[buf][(n0 + 8 + j) * LDB + kr] = d1.s[j];
    }
  };

  f32x8 c[2][4] = {};
  stageA(0, 0);
  stageB(0, 0);

  const int nk = K >> 5;
  for (int ik = 0; ik < nk; ++ik) {
    const int cur = ik & 1;
    wait_async();              // my async stores into buf 'cur' are done
    __syncthreads();           // everyone's stores visible, prev reads done
    if (ik + 1 < nk) {         // overlap next tile staging with compute
      stageA(cur ^ 1, (ik + 1) << 5);
      stageB(cur ^ 1, (ik + 1) << 5);
    }

    FragBF a[2], b[4];
#pragma unroll
    for (int tm = 0; tm < 2; ++tm) {   // A frag: e -> K = lh*8+e / 16+lh*8+(e-8)
      const int base = (wm * 32 + tm * 16 + l16) * LDA + lh * 8;
      a[tm].u[0] = *(const uint4*)(&lA[cur][base]);
      a[tm].u[1] = *(const uint4*)(&lA[cur][base + 16]);
    }
#pragma unroll
    for (int tn = 0; tn < 4; ++tn) {   // B frag: e -> K = lh*8+e / 16+lh*8+(e-8)
      const int base = (wn * 64 + tn * 16 + l16) * LDB + lh * 8;
      b[tn].u[0] = *(const uint4*)(&lB[cur][base]);
      b[tn].u[1] = *(const uint4*)(&lB[cur][base + 16]);
    }
#pragma unroll
    for (int tm = 0; tm < 2; ++tm)
#pragma unroll
      for (int tn = 0; tn < 4; ++tn)
        c[tm][tn] = wmma_bf16(a[tm].v, b[tn].v, c[tm][tn]);
  }

  // epilogue: C frag row map M = r + 8*lh, col = l16
#pragma unroll
  for (int tm = 0; tm < 2; ++tm) {
#pragma unroll
    for (int tn = 0; tn < 4; ++tn) {
      const int gn = bn0 + wn * 64 + tn * 16 + l16;
      const float bv = bias[gn];
#pragma unroll
      for (int r = 0; r < 8; ++r) {
        const int gm = bm0 + wm * 32 + tm * 16 + lh * 8 + r;
        float v = c[tm][tn][r] + bv;
        if (relu) v = fmaxf(v, 0.f);
        const size_t off = (size_t)gm * N + gn;
        if (resid) v += resid[off];
        if (outF) outF[off] = v;
        if (outB) outB[off] = f2bf(v);
      }
    }
  }
}

// ---------------------------------------------------------------------------
// Flash attention, bf16 WMMA.  qkv layout: (B,S, H*192) with head h:
// q at col h*192, k at h*192+64, v at h*192+128.
// Grid: (B*H, S/128); 8 waves, each owns 16 query rows.
// K tile staged async (straight copy); V needs a transpose -> regular path.
// Replicates reference masking: mask==1 -> score = -1e9 (post-scale).
// ---------------------------------------------------------------------------
#define LDK 80
#define LDV 40
#define LDP 40

__global__ __launch_bounds__(256)
void attention_kernel(const u16* __restrict__ qkv, const int* __restrict__ mask,
                      u16* __restrict__ att) {
  __shared__ u16 lK[32 * LDK];
  __shared__ u16 lV[64 * LDV];
  __shared__ u16 lP[8][16 * LDP];
  const int bh = blockIdx.x, b = bh / Hn, h = bh % Hn;
  const int tid = threadIdx.x;
  const int wid = tid >> 5, lane = tid & 31;
  const int lh  = lane >> 4, l16 = lane & 15;
  const int q0  = blockIdx.y * 128 + wid * 16;

  // Q fragments pre-scaled by 1/sqrt(HD)
  FragBF qf[2];
  {
    const u16* qr = qkv + (size_t)(b * Ssz + q0 + l16) * (Hn * 192) + h * 192;
    qf[0].u[0] = *(const uint4*)(qr + lh * 8);
    qf[0].u[1] = *(const uint4*)(qr + 16 + lh * 8);
    qf[1].u[0] = *(const uint4*)(qr + 32 + lh * 8);
    qf[1].u[1] = *(const uint4*)(qr + 48 + lh * 8);
#pragma unroll
    for (int f = 0; f < 2; ++f) {
      union { bf16x16 v; u16 s[16]; } t; t.v = qf[f].v;
#pragma unroll
      for (int e = 0; e < 16; ++e) t.s[e] = f2bf(bf2f(t.s[e]) * 0.125f);
      qf[f].v = t.v;
    }
  }

  float mrow[8], lrow[8];
  f32x8 o[4] = {};
#pragma unroll
  for (int r = 0; r < 8; ++r) { mrow[r] = -3.0e38f; lrow[r] = 0.f; }

  for (int kt = 0; kt < Ssz; kt += 32) {
    __syncthreads();   // previous tile fully consumed
    {   // stage K tile async (row-major == B-fragment layout for Q.K^T)
      const int key = tid >> 3, part = tid & 7;
      async_copy_b128(lds_addr(&lK[key * LDK + part * 8]),
                      qkv + (size_t)(b * Ssz + kt + key) * (Hn * 192)
                          + h * 192 + 64 + part * 8);
    }
    {   // stage V transposed -> lV[hd][key]
      const int key = tid >> 3, hd0 = (tid & 7) * 8;
      union { uint4 u; u16 s[8]; } d;
      d.u = *(const uint4*)(qkv + (size_t)(b * Ssz + kt + key) * (Hn * 192)
                            + h * 192 + 128 + hd0);
#pragma unroll
      for (int j = 0; j < 8; ++j) lV[(hd0 + j) * LDV + key] = d.s[j];
    }
    wait_async();
    __syncthreads();

    // scores 16 x 32
    f32x8 scf[2] = {};
#pragma unroll
    for (int sub = 0; sub < 2; ++sub) {
      const int key = sub * 16 + l16;
      FragBF k0f, k1f;
      k0f.u[0] = *(const uint4*)(lK + key * LDK + lh * 8);
      k0f.u[1] = *(const uint4*)(lK + key * LDK + lh * 8 + 16);
      k1f.u[0] = *(const uint4*)(lK + key * LDK + 32 + lh * 8);
      k1f.u[1] = *(const uint4*)(lK + key * LDK + 32 + lh * 8 + 16);
      scf[sub] = wmma_bf16(qf[0].v, k0f.v, scf[sub]);
      scf[sub] = wmma_bf16(qf[1].v, k1f.v, scf[sub]);
    }
    // reference masking: mask==1 => -1e9
#pragma unroll
    for (int sub = 0; sub < 2; ++sub) {
      const int mv = mask[b * Ssz + kt + sub * 16 + l16];
#pragma unroll
      for (int r = 0; r < 8; ++r)
        if (mv) scf[sub][r] = -1.0e9f;
    }
    // online softmax
    float fac[8];
#pragma unroll
    for (int r = 0; r < 8; ++r) {
      float v = fmaxf(scf[0][r], scf[1][r]);
#pragma unroll
      for (int off = 1; off < 16; off <<= 1) v = fmaxf(v, __shfl_xor(v, off, 32));
      float mn = fmaxf(mrow[r], v);
      fac[r] = __expf(mrow[r] - mn);
      mrow[r] = mn;
    }
    float ls[8];
#pragma unroll
    for (int r = 0; r < 8; ++r) ls[r] = 0.f;
#pragma unroll
    for (int sub = 0; sub < 2; ++sub)
#pragma unroll
      for (int r = 0; r < 8; ++r) {
        float p = __expf(scf[sub][r] - mrow[r]);
        ls[r] += p;
        lP[wid][(lh * 8 + r) * LDP + sub * 16 + l16] = f2bf(p);
      }
#pragma unroll
    for (int r = 0; r < 8; ++r) {
      float v = ls[r];
#pragma unroll
      for (int off = 1; off < 16; off <<= 1) v += __shfl_xor(v, off, 32);
      lrow[r] = lrow[r] * fac[r] + v;
    }
#pragma unroll
    for (int nt = 0; nt < 4; ++nt)
#pragma unroll
      for (int r = 0; r < 8; ++r) o[nt][r] *= fac[r];

    // P (C layout) -> A fragment via wave-local LDS (in-order per wave)
    FragBF pf;
    pf.u[0] = *(const uint4*)(&lP[wid][l16 * LDP + lh * 8]);
    pf.u[1] = *(const uint4*)(&lP[wid][l16 * LDP + 16 + lh * 8]);
#pragma unroll
    for (int nt = 0; nt < 4; ++nt) {
      const int hd = nt * 16 + l16;
      FragBF vf;
      vf.u[0] = *(const uint4*)(lV + hd * LDV + lh * 8);
      vf.u[1] = *(const uint4*)(lV + hd * LDV + lh * 8 + 16);
      o[nt] = wmma_bf16(pf.v, vf.v, o[nt]);
    }
  }

  float inv[8];
#pragma unroll
  for (int r = 0; r < 8; ++r) inv[r] = 1.f / lrow[r];
#pragma unroll
  for (int nt = 0; nt < 4; ++nt)
#pragma unroll
    for (int r = 0; r < 8; ++r) {
      const int s   = q0 + lh * 8 + r;
      const int col = h * HDm + nt * 16 + l16;
      att[(size_t)(b * Ssz + s) * Dm + col] = f2bf(o[nt][r] * inv[r]);
    }
}

// ---------------------------------------------------------------------------
// Pooled layernorm on x[:,0,:] + classifier (D=768 -> C=2)
// ---------------------------------------------------------------------------
__global__ __launch_bounds__(256)
void pooled_cls_kernel(const float* __restrict__ x, const float* __restrict__ hs,
                       const float* __restrict__ hb, const float* __restrict__ cw,
                       const float* __restrict__ cb, float* __restrict__ out) {
  __shared__ float red[8];
  const int bidx = blockIdx.x;                 // batch index
  const float* xr = x + (size_t)bidx * Ssz * Dm;   // row s = 0
  const int t = threadIdx.x;
  float v0 = xr[t], v1 = xr[t + 256], v2 = xr[t + 512];
  float mean = block_sum(v0 + v1 + v2, red) * (1.f / Dm);
  float d0 = v0 - mean, d1 = v1 - mean, d2 = v2 - mean;
  float var = block_sum(d0 * d0 + d1 * d1 + d2 * d2, red) * (1.f / Dm);
  float inv = rsqrtf(var + EPSC);
  float y0 = d0 * inv * hs[t]       + hb[t];
  float y1 = d1 * inv * hs[t + 256] + hb[t + 256];
  float y2 = d2 * inv * hs[t + 512] + hb[t + 512];
  float a0 = y0 * cw[t * 2]     + y1 * cw[(t + 256) * 2]     + y2 * cw[(t + 512) * 2];
  float a1 = y0 * cw[t * 2 + 1] + y1 * cw[(t + 256) * 2 + 1] + y2 * cw[(t + 512) * 2 + 1];
  float s0 = block_sum(a0, red);
  float s1 = block_sum(a1, red);
  if (t == 0) {
    out[bidx * 2 + 0] = s0 + cb[0];
    out[bidx * 2 + 1] = s1 + cb[1];
  }
}

// ---------------------------------------------------------------------------
// Host orchestration
// ---------------------------------------------------------------------------
extern "C" void kernel_launch(void* const* d_in, const int* in_sizes, int n_in,
                              void* d_out, int out_size, void* d_ws, size_t ws_size,
                              hipStream_t stream) {
  (void)in_sizes; (void)n_in; (void)out_size; (void)ws_size;
  const int*   ids    = (const int*)d_in[0];
  const int*   amask  = (const int*)d_in[1];
  const float* tok    = (const float*)d_in[2];
  const float* pos    = (const float*)d_in[3];
  const float* qkv_w  = (const float*)d_in[4];
  const float* qkv_b  = (const float*)d_in[5];
  const float* out_w  = (const float*)d_in[6];
  const float* out_b  = (const float*)d_in[7];
  const float* n1_s   = (const float*)d_in[8];
  const float* n1_b   = (const float*)d_in[9];
  const float* ff1_w  = (const float*)d_in[10];
  const float* ff1_b  = (const float*)d_in[11];
  const float* ff2_w  = (const float*)d_in[12];
  const float* ff2_b  = (const float*)d_in[13];
  const float* n2_s   = (const float*)d_in[14];
  const float* n2_b   = (const float*)d_in[15];
  const float* hln_s  = (const float*)d_in[16];
  const float* hln_b  = (const float*)d_in[17];
  const float* cls_w  = (const float*)d_in[18];
  const float* cls_b  = (const float*)d_in[19];

  const size_t BS = (size_t)Bsz * Ssz;         // 8192 rows
  char* w = (char*)d_ws;
  float* x    = (float*)w; w += BS * Dm  * sizeof(float);       // 25.2 MB
  u16*   hbuf = (u16*)w;   w += BS * Dm  * sizeof(u16);         // 12.6 MB
  u16*   qkv  = (u16*)w;   w += BS * 3 * Dm * sizeof(u16);      // 37.7 MB
  u16*   att  = (u16*)w;   w += BS * Dm  * sizeof(u16);         // 12.6 MB
  u16*   ff   = (u16*)w;   w += BS * DFm * sizeof(u16);         // 50.3 MB
  u16*   wbuf = (u16*)w;   w += (size_t)DFm * Dm * sizeof(u16); // 4.7 MB

  embed_kernel<<<(int)BS, 256, 0, stream>>>(ids, tok, pos, x);

  for (int i = 0; i < Ln; ++i) {
    // ---- attention block ----
    layernorm_kernel<<<(int)BS, 256, 0, stream>>>(x, n1_s + i * Dm, n1_b + i * Dm, hbuf);
    convert_kernel<<<2048, 256, 0, stream>>>(qkv_w + (size_t)i * Dm * 3 * Dm, wbuf, Dm * 3 * Dm);
    gemm_bf16_kernel<<<dim3(3 * Dm / BN, (int)BS / BM), 256, 0, stream>>>(
        hbuf, wbuf, qkv_b + (size_t)i * 3 * Dm, nullptr, nullptr, qkv,
        (int)BS, 3 * Dm, Dm, 0);
    attention_kernel<<<dim3(Bsz * Hn, Ssz / 128), 256, 0, stream>>>(qkv, amask, att);
    convert_kernel<<<2048, 256, 0, stream>>>(out_w + (size_t)i * Dm * Dm, wbuf, Dm * Dm);
    gemm_bf16_kernel<<<dim3(Dm / BN, (int)BS / BM), 256, 0, stream>>>(
        att, wbuf, out_b + (size_t)i * Dm, x, x, nullptr, (int)BS, Dm, Dm, 0);
    // ---- feed-forward block ----
    layernorm_kernel<<<(int)BS, 256, 0, stream>>>(x, n2_s + i * Dm, n2_b + i * Dm, hbuf);
    convert_kernel<<<2048, 256, 0, stream>>>(ff1_w + (size_t)i * Dm * DFm, wbuf, Dm * DFm);
    gemm_bf16_kernel<<<dim3(DFm / BN, (int)BS / BM), 256, 0, stream>>>(
        hbuf, wbuf, ff1_b + (size_t)i * DFm, nullptr, nullptr, ff,
        (int)BS, DFm, Dm, 1);
    convert_kernel<<<2048, 256, 0, stream>>>(ff2_w + (size_t)i * DFm * Dm, wbuf, DFm * Dm);
    gemm_bf16_kernel<<<dim3(Dm / BN, (int)BS / BM), 256, 0, stream>>>(
        ff, wbuf, ff2_b + (size_t)i * Dm, x, x, nullptr, (int)BS, Dm, DFm, 0);
  }

  pooled_cls_kernel<<<Bsz, 256, 0, stream>>>(x, hln_s, hln_b, cls_w, cls_b,
                                             (float*)d_out);
}
